// LmLSTM_70016556860044
// MI455X (gfx1250) — matrix operand order repaired and checked by reading, not compiled, not run on hardware
//
#include <hip/hip_runtime.h>
#include <hip/hip_bf16.h>

typedef __attribute__((ext_vector_type(16))) _Float16 v16h;
typedef __attribute__((ext_vector_type(8)))  float    v8f;
typedef long long i64;

union H16 { v16h v; uint4 q[2]; };

// ---- WMMA fragment loaders ---------------------------------------------
// A-matrix 16x32 f16 layout (ISA 7.12.2): lanes 0-15 row M=lane, halves 0..7 =
// K 0..7 (+8 for lanes 16-31), halves 8..15 = K 16..23 (+8 for lanes 16-31).
__device__ __forceinline__ v16h load_a_frag(const _Float16* base, i64 rowStride, int k0) {
    const int lane = threadIdx.x & 31;
    const int r    = lane & 15;
    const int hi   = lane >> 4;
    const _Float16* p = base + (i64)r * rowStride + k0 + 8 * hi;
    H16 u;
    u.q[0] = *(const uint4*)(p);
    u.q[1] = *(const uint4*)(p + 16);
    return u.v;
}

// B fragments are pre-packed frag-linear: 32 lanes x 16 halves contiguous.
__device__ __forceinline__ v16h load_b_frag(const _Float16* frag) {
    const int lane = threadIdx.x & 31;
    const _Float16* p = frag + lane * 16;
    H16 u;
    u.q[0] = *(const uint4*)(p);
    u.q[1] = *(const uint4*)(p + 8);
    return u.v;
}

// ---- Weight f32 -> f16 fragment-linear reorder --------------------------
// B-matrix 32x16 f16 layout: lane = col (lanes 0-15), halves h: K = (h&7) + 8*(lane>=16) + 16*(h>=8).
__global__ void cvt_frag(const float* __restrict__ W, _Float16* __restrict__ dst,
                         int K, int N) {
    const int KT = K >> 5;
    const i64 total = (i64)K * N;
    for (i64 i = (i64)blockIdx.x * blockDim.x + threadIdx.x; i < total;
         i += (i64)gridDim.x * blockDim.x) {
        int h    = (int)(i & 15);
        int lane = (int)((i >> 4) & 31);
        i64 fragi = i >> 9;
        int kt = (int)(fragi % KT);
        i64 nt = fragi / KT;
        int hi = lane >> 4;
        i64 k  = (i64)kt * 32 + (h & 7) + 8 * hi + 16 * (h >> 3);
        i64 n  = nt * 16 + (lane & 15);
        dst[i] = (_Float16)W[k * N + n];
    }
}

// ---- Embedding lookup + cast to f16 -------------------------------------
__global__ void embed_kernel(const int* __restrict__ inputs, const float* __restrict__ emb,
                             _Float16* __restrict__ X, int E) {
    const int row = blockIdx.x;                 // b*T + t
    const i64 tok = inputs[row];
    for (int e = threadIdx.x; e < E; e += blockDim.x)
        X[(i64)row * E + e] = (_Float16)emb[tok * E + e];
}

// ---- Generic GEMM: C[M x N] = A[M x K](f16) @ Bfrag(f16) + bias ----------
// grid.x = M/16, grid.y = ceil(N/512). 8 waves, each wave: 16x64 tile.
// Requires N % 64 == 0 so every wave tile is fully valid or fully skipped.
// NTSTORE: use non-temporal stores (streaming output, keep weights in L2).
template <bool NTSTORE>
__global__ __launch_bounds__(256) void gemm_bias(
    const _Float16* __restrict__ A, i64 lda,
    const _Float16* __restrict__ Bf,
    const float* __restrict__ bias,
    float* __restrict__ C, i64 ldc, int N, int K) {
    const int KT   = K >> 5;
    const int wave = threadIdx.x >> 5;
    const int lane = threadIdx.x & 31;
    const int m0   = blockIdx.x * 16;
    // n0 is wave-uniform: force it scalar so bounds check is an s_cbranch,
    // keeping EXEC all-ones around every WMMA (ISA 7.12 requirement).
    const int n0   = __builtin_amdgcn_readfirstlane(blockIdx.y * 512 + wave * 64);
    if (n0 + 64 > N) return;          // whole wave out of range (N % 64 == 0)
    const int nt0  = n0 >> 4;

    const _Float16* bp0 = Bf + ((i64)(nt0 + 0) * KT << 9);
    const _Float16* bp1 = Bf + ((i64)(nt0 + 1) * KT << 9);
    const _Float16* bp2 = Bf + ((i64)(nt0 + 2) * KT << 9);
    const _Float16* bp3 = Bf + ((i64)(nt0 + 3) * KT << 9);
    const _Float16* Abase = A + (i64)m0 * lda;

    v8f acc0 = {}, acc1 = {}, acc2 = {}, acc3 = {};
    // depth-1 pipeline with fixed registers: consume, then reload in place.
    v16h a  = load_a_frag(Abase, lda, 0);
    v16h b0 = load_b_frag(bp0);
    v16h b1 = load_b_frag(bp1);
    v16h b2 = load_b_frag(bp2);
    v16h b3 = load_b_frag(bp3);
#pragma unroll 2
    for (int kt = 0; kt < KT - 1; ++kt) {
        acc0 = __builtin_amdgcn_wmma_f32_16x16x32_f16(false, a, false, b0, (short)0, acc0, false, false);
        acc1 = __builtin_amdgcn_wmma_f32_16x16x32_f16(false, a, false, b1, (short)0, acc1, false, false);
        acc2 = __builtin_amdgcn_wmma_f32_16x16x32_f16(false, a, false, b2, (short)0, acc2, false, false);
        acc3 = __builtin_amdgcn_wmma_f32_16x16x32_f16(false, a, false, b3, (short)0, acc3, false, false);
        const i64 fo = (i64)(kt + 1) << 9;
        a  = load_a_frag(Abase, lda, (kt + 1) * 32);
        b0 = load_b_frag(bp0 + fo);
        b1 = load_b_frag(bp1 + fo);
        b2 = load_b_frag(bp2 + fo);
        b3 = load_b_frag(bp3 + fo);
    }
    acc0 = __builtin_amdgcn_wmma_f32_16x16x32_f16(false, a, false, b0, (short)0, acc0, false, false);
    acc1 = __builtin_amdgcn_wmma_f32_16x16x32_f16(false, a, false, b1, (short)0, acc1, false, false);
    acc2 = __builtin_amdgcn_wmma_f32_16x16x32_f16(false, a, false, b2, (short)0, acc2, false, false);
    acc3 = __builtin_amdgcn_wmma_f32_16x16x32_f16(false, a, false, b3, (short)0, acc3, false, false);

    const int col16 = lane & 15;
    const int rbase = (lane >> 4) * 8;   // C/D layout: row = m0 + r + 8*hi
    v8f accs[4] = {acc0, acc1, acc2, acc3};
#pragma unroll
    for (int j = 0; j < 4; ++j) {
        const int n = n0 + j * 16 + col16;
        const float bv = bias[n];
        float* cp = C + (i64)(m0 + rbase) * ldc + n;
#pragma unroll
        for (int r = 0; r < 8; ++r) {
            const float v = accs[j][r] + bv;
            if (NTSTORE) __builtin_nontemporal_store(v, cp + (i64)r * ldc);
            else         cp[(i64)r * ldc] = v;
        }
    }
}

// ---- One LSTM recurrence timestep ---------------------------------------
// grid.x = H/32 blocks, 256 threads = 8 waves: wave (gate*2 + sub) computes the
// 16x16 z-tile at columns gate*H + hc0 + sub*16.  z = xz_t + h_{t-1} @ Wr.
// Only ~32 WGs run here: depth-2 pipeline via a parity FIFO in fixed registers
// (consume stage s, reload the same variables for kt+2) -> no rotation copies.
__global__ __launch_bounds__(256) void lstm_step(
    const float* __restrict__ xz,        // [B*T][4H], bias already included
    const _Float16* __restrict__ Wrf,    // frag-linear [4H x H]
    const _Float16* __restrict__ Aprev,  // h_{t-1} base (Hout + (t-1)*H) or zeros
    i64 aStride,                         // T*H, or 0 for the zero buffer
    _Float16* __restrict__ Hout,         // [B*T][H]
    float* __restrict__ c,               // [B][H]
    const int* __restrict__ inputs, int t, int T, int H) {
    __shared__ float zbuf[8][16][16];
    const int wave = __builtin_amdgcn_readfirstlane(threadIdx.x >> 5);
    const int lane = threadIdx.x & 31;
    const int hc0  = blockIdx.x * 32;
    const int gate = wave >> 1;
    const int sub  = wave & 1;
    const int zcol0 = gate * H + hc0 + sub * 16;
    const int KT = H >> 5;               // 32 k-steps (even)
    const int nt = zcol0 >> 4;
    const _Float16* bp = Wrf + ((i64)nt * KT << 9);
    v8f acc = {};
    v16h aa0 = load_a_frag(Aprev, aStride, 0);
    v16h bb0 = load_b_frag(bp);
    v16h aa1 = load_a_frag(Aprev, aStride, 32);
    v16h bb1 = load_b_frag(bp + 512);
    for (int kt = 0; kt < KT - 2; kt += 2) {
        acc = __builtin_amdgcn_wmma_f32_16x16x32_f16(
            false, aa0, false, bb0, (short)0, acc, false, false);
        aa0 = load_a_frag(Aprev, aStride, (kt + 2) * 32);
        bb0 = load_b_frag(bp + ((i64)(kt + 2) << 9));
        acc = __builtin_amdgcn_wmma_f32_16x16x32_f16(
            false, aa1, false, bb1, (short)0, acc, false, false);
        aa1 = load_a_frag(Aprev, aStride, (kt + 3) * 32);
        bb1 = load_b_frag(bp + ((i64)(kt + 3) << 9));
    }
    acc = __builtin_amdgcn_wmma_f32_16x16x32_f16(
        false, aa0, false, bb0, (short)0, acc, false, false);
    acc = __builtin_amdgcn_wmma_f32_16x16x32_f16(
        false, aa1, false, bb1, (short)0, acc, false, false);
    {
        const int col16 = lane & 15;
        const int rb    = (lane >> 4) * 8;
        const int zc    = zcol0 + col16;
#pragma unroll
        for (int r = 0; r < 8; ++r) {
            int brow = rb + r;   // batch index (M dimension)
            zbuf[wave][brow][col16] = acc[r] + xz[((i64)brow * T + t) * (i64)(4 * H) + zc];
        }
    }
    __syncthreads();
    // gates epilogue: 16 batch rows x 32 h-cols
    for (int idx = threadIdx.x; idx < 512; idx += 256) {
        int brow = idx >> 5;
        int cc   = idx & 31;
        int wsel = cc >> 4, c16 = cc & 15;
        float iv = zbuf[0 + wsel][brow][c16];
        float fv = zbuf[2 + wsel][brow][c16];
        float gv = zbuf[4 + wsel][brow][c16];
        float ov = zbuf[6 + wsel][brow][c16];
        iv = 1.f / (1.f + __expf(-iv));
        fv = 1.f / (1.f + __expf(-fv));
        gv = tanhf(gv);
        ov = 1.f / (1.f + __expf(-ov));
        int hcol = hc0 + cc;
        i64 cidx = (i64)brow * H + hcol;
        float cp = c[cidx];
        float cn = fv * cp + iv * gv;
        float hn = ov * tanhf(cn);
        bool m = inputs[brow * T + t] != 0;
        float hprev = (t == 0) ? 0.f : (float)Hout[((i64)brow * T + (t - 1)) * H + hcol];
        float h2 = m ? hn : hprev;
        c[cidx] = m ? cn : cp;
        Hout[((i64)brow * T + t) * H + hcol] = (_Float16)h2;
    }
}

// ---- Masked rows of logits become one_hot(0) -----------------------------
__global__ void mask_fixup(const int* __restrict__ inputs, float* __restrict__ out, int V) {
    const int row = blockIdx.x;
    if (inputs[row] != 0) return;
    for (int v = threadIdx.x; v < V; v += blockDim.x)
        out[(i64)row * V + v] = (v == 0) ? 1.f : 0.f;
}

extern "C" void kernel_launch(void* const* d_in, const int* in_sizes, int n_in,
                              void* d_out, int out_size, void* d_ws, size_t ws_size,
                              hipStream_t stream) {
    const int B = 16, T = 256, V = 32000, E = 512, H = 1024;
    const int   M4 = B * T;          // 4096 rows
    const int*   inputs = (const int*)d_in[0];
    const float* emb    = (const float*)d_in[1];
    const float* Wk0    = (const float*)d_in[2];
    const float* Wr0    = (const float*)d_in[3];
    const float* b0     = (const float*)d_in[4];
    const float* Wk1    = (const float*)d_in[5];
    const float* Wr1    = (const float*)d_in[6];
    const float* b1     = (const float*)d_in[7];
    const float* Wout   = (const float*)d_in[8];
    const float* bout   = (const float*)d_in[9];
    float* out = (float*)d_out;

    size_t off = 0;
    auto alloc = [&](size_t bytes) -> void* {
        off = (off + 255) & ~(size_t)255;
        void* p = (char*)d_ws + off;
        off += bytes;
        return p;
    };
    _Float16* Wk0f  = (_Float16*)alloc((size_t)E * 4 * H * 2);
    _Float16* Wr0f  = (_Float16*)alloc((size_t)H * 4 * H * 2);
    _Float16* Wk1f  = (_Float16*)alloc((size_t)H * 4 * H * 2);
    _Float16* Wr1f  = (_Float16*)alloc((size_t)H * 4 * H * 2);
    _Float16* Woutf = (_Float16*)alloc((size_t)H * V * 2);
    _Float16* X0h   = (_Float16*)alloc((size_t)M4 * E * 2);
    _Float16* H0h   = (_Float16*)alloc((size_t)M4 * H * 2);
    _Float16* H1h   = (_Float16*)alloc((size_t)M4 * H * 2);
    float*    xz0   = (float*)alloc((size_t)M4 * 4 * H * 4);
    float*    xz1   = (float*)alloc((size_t)M4 * 4 * H * 4);
    float*    c0    = (float*)alloc((size_t)B * H * 4);
    float*    c1    = (float*)alloc((size_t)B * H * 4);
    _Float16* zeroh = (_Float16*)alloc(4096);

    hipMemsetAsync(c0, 0, (size_t)B * H * 4, stream);
    hipMemsetAsync(c1, 0, (size_t)B * H * 4, stream);
    hipMemsetAsync(zeroh, 0, 4096, stream);

    // weight conversion to WMMA fragment-linear f16 (stays hot in 192MB L2)
    cvt_frag<<<2048, 256, 0, stream>>>(Wk0,  Wk0f,  E, 4 * H);
    cvt_frag<<<2048, 256, 0, stream>>>(Wr0,  Wr0f,  H, 4 * H);
    cvt_frag<<<2048, 256, 0, stream>>>(Wk1,  Wk1f,  H, 4 * H);
    cvt_frag<<<2048, 256, 0, stream>>>(Wr1,  Wr1f,  H, 4 * H);
    cvt_frag<<<4096, 256, 0, stream>>>(Wout, Woutf, H, V);

    embed_kernel<<<M4, 256, 0, stream>>>(inputs, emb, X0h, E);

    // layer 0 input projection: xz0 = X0h @ Wk0 + b0   (4096 x 4096 x 512)
    gemm_bias<false><<<dim3(M4 / 16, (4 * H + 511) / 512), 256, 0, stream>>>(
        X0h, E, Wk0f, b0, xz0, 4 * H, 4 * H, E);

    // layer 0 recurrence
    for (int t = 0; t < T; ++t) {
        const _Float16* Ap = (t == 0) ? zeroh : (H0h + (size_t)(t - 1) * H);
        i64 stride = (t == 0) ? 0 : (i64)T * H;
        lstm_step<<<H / 32, 256, 0, stream>>>(xz0, Wr0f, Ap, stride, H0h, c0,
                                              inputs, t, T, H);
    }

    // layer 1 input projection: xz1 = H0h @ Wk1 + b1   (4096 x 4096 x 1024)
    gemm_bias<false><<<dim3(M4 / 16, (4 * H + 511) / 512), 256, 0, stream>>>(
        H0h, H, Wk1f, b1, xz1, 4 * H, 4 * H, H);

    // layer 1 recurrence
    for (int t = 0; t < T; ++t) {
        const _Float16* Ap = (t == 0) ? zeroh : (H1h + (size_t)(t - 1) * H);
        i64 stride = (t == 0) ? 0 : (i64)T * H;
        lstm_step<<<H / 32, 256, 0, stream>>>(xz1, Wr1f, Ap, stride, H1h, c1,
                                              inputs, t, T, H);
    }

    // logits = H1h @ W_out + b_out   (4096 x 32000 x 1024)
    // streaming 524MB output: non-temporal stores keep W_out resident in L2
    gemm_bias<true><<<dim3(M4 / 16, (V + 511) / 512), 256, 0, stream>>>(
        H1h, H, Woutf, bout, out, V, V, H);

    mask_fixup<<<M4, 256, 0, stream>>>(inputs, out, V);
}
